// MinCutGAT_Sparse_12300786336173
// MI455X (gfx1250) — matrix operand order repaired and checked by reading, not compiled, not run on hardware
//
#include <hip/hip_runtime.h>
#include <hip/hip_bf16.h>

#define N_NODES 200000
#define E_EDGES 6400000
#define DIN     256
#define KC      30      // clusters
#define HEADS   4
#define DH      64

// output offsets (floats): out(960) | mincut(1) | ortho(1) | attn(900) | Z(7680) | S(6,000,000)
#define OFF_MINCUT 960
#define OFF_ORTHO  961
#define OFF_ATTN   962
#define OFF_Z      1862
#define OFF_S      9542

typedef __attribute__((ext_vector_type(16))) __bf16 v16bf;
typedef __attribute__((ext_vector_type(8)))  float  v8f;
typedef __attribute__((ext_vector_type(4)))  unsigned int v4u;
typedef __attribute__((ext_vector_type(8)))  int    v8i;
typedef __attribute__((ext_vector_type(4)))  int    v4i;

__device__ __forceinline__ v8f vzero8() {
  v8f z;
#pragma unroll
  for (int i = 0; i < 8; ++i) z[i] = 0.f;
  return z;
}

__device__ __forceinline__ v8f wmma_bf(v16bf a, v16bf b, v8f c) {
  return __builtin_amdgcn_wmma_f32_16x16x32_bf16(false, a, false, b, (short)0, c, false, false);
}

// A fragment (16x32, M=row per lane, K halves by lane group) from f32 row pointer (row base + k0)
__device__ __forceinline__ v16bf afrag_f32(const float* rowp, int lane) {
  int koff = ((lane >> 4) & 1) * 16;
  v16bf a;
#pragma unroll
  for (int j = 0; j < 8; ++j) {
    a[j]     = (__bf16)rowp[koff + j];
    a[j + 8] = (__bf16)rowp[koff + 8 + j];
  }
  return a;
}

__device__ __forceinline__ v16bf afrag_bf(const __bf16* rowp, int lane) {
  int koff = ((lane >> 4) & 1) * 16;
  v16bf a;
#pragma unroll
  for (int j = 0; j < 8; ++j) {
    a[j]     = rowp[koff + j];
    a[j + 8] = rowp[koff + 8 + j];
  }
  return a;
}

// B fragment: column n fixed per lane, 16 contiguous K values starting at p
__device__ __forceinline__ v16bf bfrag_contig(const __bf16* p) {
  v16bf b;
#pragma unroll
  for (int j = 0; j < 16; ++j) b[j] = p[j];
  return b;
}

// ---- Tensor Data Mover: 1D copy of n_elem4 dwords global->LDS (D# per ISA ch.8) ----
__device__ __forceinline__ void tdm_load_dwords(unsigned ldsOff, unsigned long long ga, int n_elem4) {
  v4u g0;
  g0[0] = 1u;                                          // count=1, user descriptor
  g0[1] = ldsOff;                                      // lds_addr
  g0[2] = (unsigned)ga;                                // global_addr lo
  g0[3] = (unsigned)(ga >> 32) | 0x80000000u;          // global_addr hi | type=2
  v8i g1;
  g1[0] = 0x00020000;                                  // data_size=4B
  g1[1] = (int)(((unsigned)n_elem4 & 0xFFFFu) << 16);  // tensor_dim0 lo16
  g1[2] = (int)(((unsigned)n_elem4 >> 16) | (1u << 16)); // tensor_dim0 hi | tensor_dim1=1
  g1[3] = (int)(((unsigned)n_elem4 & 0xFFFFu) << 16);  // tile_dim0
  g1[4] = 1;                                           // tile_dim1=1, tile_dim2=0
  g1[5] = n_elem4;                                     // tensor_dim0_stride lo32
  g1[6] = 0;
  g1[7] = 0;
  v4i gz; gz[0] = gz[1] = gz[2] = gz[3] = 0;
#if __clang_major__ >= 23
  v8i gz8;
#pragma unroll
  for (int i = 0; i < 8; ++i) gz8[i] = 0;
  __builtin_amdgcn_tensor_load_to_lds(g0, g1, gz, gz, gz8, 0);
#else
  __builtin_amdgcn_tensor_load_to_lds(g0, g1, gz, gz, 0);
#endif
}

// =====================================================================================
// Kernel 1: S = softmax(x@assign_W+b); writes S; accumulates zacc=S^T x (bf16 WMMA),
//           ssacc=S^T S, colsum(S). x tiles double-buffered into LDS via TDM so the
//           DMA for block i+1 overlaps WMMA/softmax on block i (s_wait_tensorcnt 1).
// =====================================================================================
__global__ void __launch_bounds__(256) k_assign(const float* __restrict__ x,
                                                const float* __restrict__ aW,
                                                const float* __restrict__ ab,
                                                float* __restrict__ S_out,
                                                float* __restrict__ zacc,
                                                float* __restrict__ ssacc,
                                                float* __restrict__ colsum) {
  __shared__ float  xs[2][32 * 256];  // double-buffered x block (TDM dest)
  __shared__ __bf16 xbT[256 * 32];    // x block transposed bf16 [feat][node]
  __shared__ __bf16 wabT[32 * 256];   // assign_W^T bf16, padded to 32 cols
  __shared__ __bf16 ST[32 * 32];      // S^T bf16 [cluster][node]
  __shared__ float  logit[32 * 32];
  __shared__ float  cs[32];
  __shared__ float  abl[32];

  const int t = threadIdx.x, lane = t & 31, wv = t >> 5;
  const int li = lane & 15, hi = (lane >> 4) & 1;

  for (int idx = t; idx < 32 * 256; idx += 256) {
    int n = idx >> 8, k = idx & 255;
    wabT[idx] = (n < KC) ? (__bf16)aW[k * KC + n] : (__bf16)0.f;
  }
  if (t < 32) {
    cs[t]  = 0.f;
    abl[t] = (t < KC) ? ab[t] : 0.f;
  }

  v8f zac[4];
#pragma unroll
  for (int q = 0; q < 4; ++q) zac[q] = vzero8();
  v8f ssa = vzero8();

  const int NB = N_NODES / 32;
  int nb = 0;
  if ((int)blockIdx.x < NB) nb = (NB - (int)blockIdx.x + (int)gridDim.x - 1) / (int)gridDim.x;

  // prologue: kick off DMA for first block
  if (nb > 0 && wv == 0) {
    tdm_load_dwords((unsigned)(size_t)&xs[0][0],
                    (unsigned long long)(size_t)(x + (size_t)blockIdx.x * 32 * DIN), 32 * 256);
  }
  __syncthreads();

  for (int i = 0; i < nb; ++i) {
    const int blk = (int)blockIdx.x + i * (int)gridDim.x;
    const int cur = i & 1;
    const int base = blk * 32;

    if (wv == 0) {
      if (i + 1 < nb) {
        const int nblk = blk + (int)gridDim.x;
        // issue next DMA into the other buffer, then wait only for the *current* one
        tdm_load_dwords((unsigned)(size_t)&xs[cur ^ 1][0],
                        (unsigned long long)(size_t)(x + (size_t)nblk * 32 * DIN), 32 * 256);
        __builtin_amdgcn_s_wait_tensorcnt(1);
      } else {
        __builtin_amdgcn_s_wait_tensorcnt(0);
      }
    }
    __syncthreads();

    const float* xc = &xs[cur][0];

    // build transposed bf16 copy for S^T x B-fragments
    {
      int nd = t >> 3;
      int f0 = (t & 7) * 32;
#pragma unroll
      for (int j = 0; j < 32; ++j) xbT[(f0 + j) * 32 + nd] = (__bf16)xc[nd * 256 + f0 + j];
    }
    __syncthreads();

    // logits = x_block @ assign_W  (four 16x16 tiles, waves 0..3)
    if (wv < 4) {
      const int mrow = (wv >> 1) * 16, ncol = (wv & 1) * 16;
      v8f acc = vzero8();
#pragma unroll
      for (int kb = 0; kb < 8; ++kb) {
        v16bf a = afrag_f32(xc + (mrow + li) * 256 + kb * 32, lane);
        v16bf b = bfrag_contig(wabT + (ncol + li) * 256 + kb * 32 + hi * 16);
        acc = wmma_bf(a, b, acc);
      }
#pragma unroll
      for (int i8 = 0; i8 < 8; ++i8) logit[(mrow + i8 + 8 * hi) * 32 + ncol + li] = acc[i8];
    }
    __syncthreads();

    // row softmax over 30 clusters; write S (f32, output) + S^T (bf16, LDS)
    if (t < 32) {
      float l[KC];
      float m = -1e30f;
      for (int k = 0; k < KC; ++k) { float v = logit[t * 32 + k] + abl[k]; l[k] = v; m = v > m ? v : m; }
      float s = 0.f;
      for (int k = 0; k < KC; ++k) { float e = __expf(l[k] - m); l[k] = e; s += e; }
      float inv = 1.f / s;
      size_t node = (size_t)(base + t);
      for (int k = 0; k < KC; ++k) {
        float p = l[k] * inv;
        S_out[node * KC + k] = p;
        ST[k * 32 + t] = (__bf16)p;
        atomicAdd(&cs[k], p);
      }
      ST[30 * 32 + t] = (__bf16)0.f;
      ST[31 * 32 + t] = (__bf16)0.f;
    }
    __syncthreads();

    // S^T fragments (clusters x nodes), shared by Z and SS accumulation
    v16bf sa0 = afrag_bf(ST + (0 + li) * 32, lane);
    v16bf sa1 = afrag_bf(ST + (16 + li) * 32, lane);

    // zacc += S^T @ x : wave w owns feature columns [32w, 32w+32)
#pragma unroll
    for (int q = 0; q < 4; ++q) {
      int mt = q >> 1, nt2 = q & 1;
      int fcol = wv * 32 + nt2 * 16 + li;
      v16bf b = bfrag_contig(xbT + fcol * 32 + hi * 16);
      zac[q] = wmma_bf(mt ? sa1 : sa0, b, zac[q]);
    }
    // ssacc += S^T @ S : waves 4..7 own one 16x16 tile each
    if (wv >= 4) {
      int idx = wv - 4, mt = idx >> 1, nt = idx & 1;
      v16bf b = bfrag_contig(ST + (nt * 16 + li) * 32 + hi * 16);
      ssa = wmma_bf(mt ? sa1 : sa0, b, ssa);
    }
    __syncthreads();
  }

  // flush accumulators
#pragma unroll
  for (int q = 0; q < 4; ++q) {
    int mt = q >> 1, nt2 = q & 1;
#pragma unroll
    for (int i = 0; i < 8; ++i) {
      int row = mt * 16 + i + 8 * hi;
      int col = wv * 32 + nt2 * 16 + li;
      atomicAdd(&zacc[row * 256 + col], zac[q][i]);
    }
  }
  if (wv >= 4) {
    int idx = wv - 4, mt = idx >> 1, nt = idx & 1;
#pragma unroll
    for (int i = 0; i < 8; ++i)
      atomicAdd(&ssacc[(mt * 16 + i + 8 * hi) * 32 + nt * 16 + li], ssa[i]);
  }
  __syncthreads();
  if (t < 32) atomicAdd(&colsum[t], cs[t]);
}

// =====================================================================================
// Kernel 2: cut = sum_e dot(S[row],S[col]); vol = sum_e rowsum(S[row]).  L2-resident.
// =====================================================================================
__global__ void __launch_bounds__(256) k_edges(const long long* __restrict__ ei,
                                               const float* __restrict__ S,
                                               float* __restrict__ scalars) {
  __shared__ float rc[256], rv[256];
  long long e = (long long)blockIdx.x * 256 + threadIdx.x;
  float cut = 0.f, vol = 0.f;
  if (e < (long long)E_EDGES) {
    long long r = ei[e];
    long long c = ei[(long long)E_EDGES + e];
    const float* a = S + r * (long long)KC;
    const float* b = S + c * (long long)KC;
#pragma unroll
    for (int k = 0; k < KC; ++k) { float av = a[k]; cut += av * b[k]; vol += av; }
  }
  rc[threadIdx.x] = cut; rv[threadIdx.x] = vol;
  __syncthreads();
  for (int s2 = 128; s2 > 0; s2 >>= 1) {
    if (threadIdx.x < (unsigned)s2) {
      rc[threadIdx.x] += rc[threadIdx.x + s2];
      rv[threadIdx.x] += rv[threadIdx.x + s2];
    }
    __syncthreads();
  }
  if (threadIdx.x == 0) { atomicAdd(&scalars[0], rc[0]); atomicAdd(&scalars[1], rv[0]); }
}

// =====================================================================================
// Kernel 3 (single WG): Z = zacc@proj_W + colsum⊗proj_b; two dense GAT layers; final
// linear; losses; attention. All matmuls as 16x16x32 bf16 WMMA, rows 30/31 zero-padded.
// =====================================================================================
__global__ void __launch_bounds__(256) k_final(const float* __restrict__ zacc,
                                               const float* __restrict__ ssacc,
                                               const float* __restrict__ colsum,
                                               const float* __restrict__ scalars,
                                               const float* __restrict__ pW,
                                               const float* __restrict__ pb,
                                               const float* __restrict__ g1W,
                                               const float* __restrict__ g1a,
                                               const float* __restrict__ g2W,
                                               const float* __restrict__ g2a,
                                               const float* __restrict__ lW,
                                               const float* __restrict__ lb,
                                               float* __restrict__ out) {
  __shared__ __bf16 bufA[32 * 256];
  __shared__ __bf16 bufB[32 * 256];
  __shared__ float  Whf[4 * 32 * 64];
  __shared__ __bf16 Whb[4 * 32 * 64];
  __shared__ __bf16 alphab[4 * 32 * 32];
  __shared__ float  srcl[4 * 32], dstl[4 * 32];
  __shared__ float  attnsum[32 * 32];
  __shared__ float  csl[32];
  __shared__ float  onorm;

  const int t = threadIdx.x, lane = t & 31, wv = t >> 5;
  const int li = lane & 15, hi = (lane >> 4) & 1;

  for (int idx = t; idx < 32 * 256; idx += 256) bufB[idx] = (__bf16)zacc[idx];
  for (int idx = t; idx < 1024; idx += 256) attnsum[idx] = 0.f;
  if (t < 32) csl[t] = colsum[t];
  if (t == 0) onorm = 0.f;
  __syncthreads();

  // ---- Z = Za @ proj_W + colsum⊗proj_b -> bufA (bf16) + output ----
#pragma unroll
  for (int q = 0; q < 4; ++q) {
    int mt = q >> 1, nt = q & 1;
    int ncol = wv * 32 + nt * 16 + li;
    v8f acc = vzero8();
    for (int kb = 0; kb < 8; ++kb) {
      v16bf a = afrag_bf(bufB + (mt * 16 + li) * 256 + kb * 32, lane);
      int ks = kb * 32 + hi * 16;
      v16bf b;
#pragma unroll
      for (int j = 0; j < 16; ++j) b[j] = (__bf16)pW[(ks + j) * 256 + ncol];
      acc = wmma_bf(a, b, acc);
    }
    float pbn = pb[ncol];
#pragma unroll
    for (int i = 0; i < 8; ++i) {
      int m = mt * 16 + i + 8 * hi;
      float v = acc[i] + csl[m] * pbn;
      if (m < KC) out[OFF_Z + m * 256 + ncol] = v;
      bufA[m * 256 + ncol] = (__bf16)v;
    }
  }
  __syncthreads();

  // ---- two GAT layers ----
  const float* gW[2] = {g1W, g2W};
  const float* ga[2] = {g1a, g2a};
  __bf16* inb = bufA;
  __bf16* onb = bufB;
  for (int layer = 0; layer < 2; ++layer) {
    const float* W  = gW[layer];
    const float* av = ga[layer];

    // Wh[h] = in @ W[h]   (32 tile-jobs over 8 waves)
#pragma unroll
    for (int jj = 0; jj < 4; ++jj) {
      int job = wv * 4 + jj;
      int h = job >> 3, mt = (job >> 2) & 1, nt = job & 3;
      int ocol = nt * 16 + li;
      v8f acc = vzero8();
      for (int kb = 0; kb < 8; ++kb) {
        v16bf a = afrag_bf(inb + (mt * 16 + li) * 256 + kb * 32, lane);
        int ks = kb * 32 + hi * 16;
        v16bf b;
#pragma unroll
        for (int j = 0; j < 16; ++j) b[j] = (__bf16)W[(h * 256 + ks + j) * 64 + ocol];
        acc = wmma_bf(a, b, acc);
      }
#pragma unroll
      for (int i = 0; i < 8; ++i) {
        int m = mt * 16 + i + 8 * hi;
        Whf[(h * 32 + m) * 64 + ocol] = acc[i];
        Whb[(h * 32 + m) * 64 + ocol] = (__bf16)acc[i];
      }
    }
    __syncthreads();

    // src/dst projections
    if (t < 128) {
      int h = t >> 5, n = t & 31;
      const float* as = av + h * 128;
      float s1 = 0.f, s2 = 0.f;
      for (int o = 0; o < 64; ++o) {
        float w = Whf[(h * 32 + n) * 64 + o];
        s1 += w * as[o];
        s2 += w * as[64 + o];
      }
      srcl[h * 32 + n] = s1;
      dstl[h * 32 + n] = s2;
    }
    __syncthreads();

    // dense attention softmax rows
    if (t < 128) {
      int h = t >> 5, i = t & 31;
      if (i < KC) {
        float ev[KC];
        float m = -1e30f;
        float si = srcl[h * 32 + i];
        for (int j = 0; j < KC; ++j) {
          float e = si + dstl[h * 32 + j];
          e = e > 0.f ? e : 0.2f * e;
          ev[j] = e;
          if (e > m) m = e;
        }
        float s = 0.f;
        for (int j = 0; j < KC; ++j) { float e = __expf(ev[j] - m); ev[j] = e; s += e; }
        float inv = 1.f / s;
        for (int j = 0; j < KC; ++j) {
          float al = ev[j] * inv;
          alphab[(h * 32 + i) * 32 + j] = (__bf16)al;
          atomicAdd(&attnsum[i * 32 + j], al);
        }
        alphab[(h * 32 + i) * 32 + 30] = (__bf16)0.f;
        alphab[(h * 32 + i) * 32 + 31] = (__bf16)0.f;
      } else {
        for (int j = 0; j < 32; ++j) alphab[(h * 32 + i) * 32 + j] = (__bf16)0.f;
      }
    }
    __syncthreads();

    // hp = alpha @ Wh, relu, concat heads -> onb
#pragma unroll
    for (int jj = 0; jj < 4; ++jj) {
      int job = wv * 4 + jj;
      int h = job >> 3, mt = (job >> 2) & 1, nt = job & 3;
      int ocol = nt * 16 + li;
      v16bf a = afrag_bf(alphab + (h * 32 + mt * 16 + li) * 32, lane);
      v16bf b;
#pragma unroll
      for (int j = 0; j < 16; ++j) b[j] = Whb[(h * 32 + hi * 16 + j) * 64 + ocol];
      v8f acc = wmma_bf(a, b, vzero8());
#pragma unroll
      for (int i = 0; i < 8; ++i) {
        int m = mt * 16 + i + 8 * hi;
        float v = acc[i];
        v = v > 0.f ? v : 0.f;
        onb[m * 256 + h * 64 + ocol] = (__bf16)v;
      }
    }
    __syncthreads();
    __bf16* tmp = inb; inb = onb; onb = tmp;
  }

  // ---- out = x2 @ lin_W + lin_b ----
  if (wv < 4) {
    int mt = wv >> 1, nt = wv & 1;
    int ncol = nt * 16 + li;
    v8f acc = vzero8();
    for (int kb = 0; kb < 8; ++kb) {
      v16bf a = afrag_bf(inb + (mt * 16 + li) * 256 + kb * 32, lane);
      int ks = kb * 32 + hi * 16;
      v16bf b;
#pragma unroll
      for (int j = 0; j < 16; ++j) b[j] = (__bf16)lW[(ks + j) * 32 + ncol];
      acc = wmma_bf(a, b, acc);
    }
    float bn = lb[ncol];
#pragma unroll
    for (int i = 0; i < 8; ++i) {
      int m = mt * 16 + i + 8 * hi;
      if (m < KC) out[m * 32 + ncol] = acc[i] + bn;
    }
  }
  __syncthreads();

  // ---- losses + attention output ----
  {
    float part = 0.f;
    for (int idx = t; idx < 1024; idx += 256) {
      int i = idx >> 5, j = idx & 31;
      if (i < KC && j < KC) {
        float d = ssacc[idx] - (i == j ? 1.f : 0.f);
        part += d * d;
      }
    }
    atomicAdd(&onorm, part);
  }
  __syncthreads();
  if (t == 0) {
    out[OFF_MINCUT] = -scalars[0] / (scalars[1] + 1e-9f);
    out[OFF_ORTHO]  = sqrtf(onorm);
  }
  for (int idx = t; idx < 900; idx += 256) {
    int i = idx / 30, j = idx % 30;
    out[OFF_ATTN + idx] = attnsum[i * 32 + j] * 0.125f;  // (sum over 4 heads x 2 layers)/8
  }
}

extern "C" void kernel_launch(void* const* d_in, const int* in_sizes, int n_in,
                              void* d_out, int out_size, void* d_ws, size_t ws_size,
                              hipStream_t stream) {
  (void)in_sizes; (void)n_in; (void)out_size; (void)ws_size;
  const float*     x   = (const float*)d_in[0];
  const long long* ei  = (const long long*)d_in[1];
  const float*     aW  = (const float*)d_in[2];
  const float*     ab  = (const float*)d_in[3];
  const float*     pW  = (const float*)d_in[4];
  const float*     pb  = (const float*)d_in[5];
  const float*     g1W = (const float*)d_in[6];
  const float*     g1a = (const float*)d_in[7];
  const float*     g2W = (const float*)d_in[8];
  const float*     g2a = (const float*)d_in[9];
  const float*     lW  = (const float*)d_in[10];
  const float*     lb  = (const float*)d_in[11];
  float* out = (float*)d_out;
  float* ws  = (float*)d_ws;

  float* zacc    = ws;          // 32*256
  float* ssacc   = ws + 8192;   // 32*32
  float* colsum  = ws + 9216;   // 32
  float* scalars = ws + 9248;   // cut, vol

  (void)hipMemsetAsync(d_ws, 0, (size_t)(9248 + 16) * sizeof(float), stream);
  k_assign<<<512, 256, 0, stream>>>(x, aW, ab, out + OFF_S, zacc, ssacc, colsum);
  k_edges<<<(E_EDGES + 255) / 256, 256, 0, stream>>>(ei, out + OFF_S, scalars);
  k_final<<<1, 256, 0, stream>>>(zacc, ssacc, colsum, scalars, pW, pb,
                                 g1W, g1a, g2W, g2a, lW, lb, out);
}
